// HomoGAT_37709812859000
// MI455X (gfx1250) — matrix operand order repaired and checked by reading, not compiled, not run on hardware
//
#include <hip/hip_runtime.h>
#include <hip/hip_bf16.h>
#include <cstdint>

// ---------------------------------------------------------------------------
// HomoGAT forward for MI455X (gfx1250): bf16 WMMA GEMMs (async-LDS double
// buffered) + scalar graph kernels
// ---------------------------------------------------------------------------

typedef __attribute__((ext_vector_type(16))) __bf16 v16bf;
typedef __attribute__((ext_vector_type(8)))  __bf16 v8bf;
typedef __attribute__((ext_vector_type(8)))  float  v8f;

#define DEV_INLINE __device__ __forceinline__

typedef __attribute__((address_space(3))) __bf16* lds_bf16_p;

DEV_INLINE unsigned lds_addr_of(__bf16* p) {
    return (unsigned)(uintptr_t)(lds_bf16_p)p;   // raw LDS byte offset
}

// async global->LDS copy, GVS mode: mem = saddr + (i32)voffset ; lds = ldsoff
DEV_INLINE void async_b128(unsigned ldsoff, int voff, const __bf16* base) {
    asm volatile("global_load_async_to_lds_b128 %0, %1, %2"
                 :: "v"(ldsoff), "v"(voff), "s"(base)
                 : "memory");
}
DEV_INLINE void wait_async0() {
    asm volatile("s_wait_asynccnt 0" ::: "memory");
}

DEV_INLINE v8f v8f_zero() {
    v8f z;
#pragma unroll
    for (int i = 0; i < 8; ++i) z[i] = 0.0f;
    return z;
}

// monotonic float<->uint mapping for atomicMax on floats
DEV_INLINE unsigned fkey(float f) {
    unsigned u = __float_as_uint(f);
    return (u & 0x80000000u) ? ~u : (u | 0x80000000u);
}
DEV_INLINE float unfkey(unsigned k) {
    return (k & 0x80000000u) ? __uint_as_float(k & 0x7fffffffu)
                             : __uint_as_float(~k);
}

// ---------------------------------------------------------------------------
// Utility kernels
// ---------------------------------------------------------------------------
__global__ void k_fill_f32(float* __restrict__ p, float v, size_t n) {
    size_t i = (size_t)blockIdx.x * blockDim.x + threadIdx.x;
    size_t st = (size_t)gridDim.x * blockDim.x;
    for (; i < n; i += st) p[i] = v;
}

__global__ void k_fill_u32(unsigned* __restrict__ p, unsigned v, size_t n) {
    size_t i = (size_t)blockIdx.x * blockDim.x + threadIdx.x;
    size_t st = (size_t)gridDim.x * blockDim.x;
    for (; i < n; i += st) p[i] = v;
}

__global__ void k_cvt_bf16(const float* __restrict__ x, __bf16* __restrict__ o, size_t n) {
    size_t i = (size_t)blockIdx.x * blockDim.x + threadIdx.x;
    size_t st = (size_t)gridDim.x * blockDim.x;
    for (; i < n; i += st) o[i] = (__bf16)x[i];
}

// W [K,F] f32 row-major -> Bt [Np][Kp] bf16 (Bt[n][k] = W[k][n]), zero padded
__global__ void k_wt(const float* __restrict__ W, __bf16* __restrict__ Bt,
                     int K, int F, int Kp, int Np) {
    size_t tot = (size_t)Np * Kp;
    size_t i = (size_t)blockIdx.x * blockDim.x + threadIdx.x;
    size_t st = (size_t)gridDim.x * blockDim.x;
    for (; i < tot; i += st) {
        int n = (int)(i / Kp);
        int k = (int)(i % Kp);
        float v = (k < K && n < F) ? W[(size_t)k * F + n] : 0.0f;
        Bt[i] = (__bf16)v;
    }
}

// out[d] = sum_f W[d*F+f] * a[f]   (tiny: D<=256)
__global__ void k_wvec(const float* __restrict__ W, const float* __restrict__ a,
                       float* __restrict__ o, int D, int F) {
    int d = blockIdx.x * blockDim.x + threadIdx.x;
    if (d >= D) return;
    float s = 0.0f;
    for (int f = 0; f < F; ++f) s += W[(size_t)d * F + f] * a[f];
    o[d] = s;
}

// per-row dual dot products: o1[n] = X[n,:].v1 ; o2[n] = X[n,:].v2
__global__ void k_rowdot2(const __bf16* __restrict__ X,
                          const float* __restrict__ v1, const float* __restrict__ v2,
                          float* __restrict__ o1, float* __restrict__ o2,
                          int Nn, int F) {
    int n = blockIdx.x * blockDim.x + threadIdx.x;
    if (n >= Nn) return;
    const __bf16* row = X + (size_t)n * F;
    float s1 = 0.0f, s2 = 0.0f;
    for (int f = 0; f < F; ++f) {
        float x = (float)row[f];
        s1 += x * v1[f];
        s2 += x * v2[f];
    }
    o1[n] = s1;
    o2[n] = s2;
}

// ---------------------------------------------------------------------------
// WMMA GEMM: C[M,ldc](f32) = A[M,Kp](bf16) @ Bt[Np][Kp]^T (bf16) (+bias)
// Block tile 128x128, 8 waves (4x2), wave tile 32x64 = 2x4 WMMA 16x16x32.
// Global->LDS via async DMA, double buffered, one barrier per K-step.
// ---------------------------------------------------------------------------
#define GEMM_BM 128
#define GEMM_BN 128
#define GEMM_BK 32
#define LDS_PAD 40  // row stride in bf16 elems: 80B = 20 banks -> conflict-free
#define BUF_BYTES (GEMM_BM * LDS_PAD * 2)

__global__ __launch_bounds__(256)
void k_gemm_bf16(const __bf16* __restrict__ A, const __bf16* __restrict__ Bt,
                 float* __restrict__ C, const float* __restrict__ bias,
                 int M, int Nvalid, int Kp, int ldc) {
    __shared__ __bf16 As[2][GEMM_BM][LDS_PAD];
    __shared__ __bf16 Bs[2][GEMM_BN][LDS_PAD];

    const int tid  = threadIdx.x;
    const int lane = tid & 31;
    const int wave = tid >> 5;    // 0..7
    const int wm   = wave >> 1;   // 0..3 : 32-row band
    const int wn   = wave & 1;    // 0..1 : 64-col band
    const int bm0  = blockIdx.y * GEMM_BM;
    const int bn0  = blockIdx.x * GEMM_BN;
    const int half = lane >> 4;   // K-half selector
    const int l15  = lane & 15;

    // per-thread staging descriptors: 2 x 16B chunks of A, 2 of B per K-step
    unsigned ldsA0[2], ldsB0[2];
    int voffA[2], voffB[2];
#pragma unroll
    for (int it = 0; it < 2; ++it) {
        int chunk = tid + it * 256;            // 0..511
        int r  = chunk >> 2;                   // 0..127
        int c8 = (chunk & 3) * 8;              // 0,8,16,24
        int gr = bm0 + r; if (gr > M - 1) gr = M - 1;   // clamp (rows >= M unused)
        ldsA0[it] = lds_addr_of(&As[0][r][c8]);
        voffA[it] = (int)(((size_t)gr * Kp + c8) * 2);
        int gn = bn0 + r;                      // Bt rows pre-padded, always valid
        ldsB0[it] = lds_addr_of(&Bs[0][r][c8]);
        voffB[it] = (int)(((size_t)gn * Kp + c8) * 2);
    }

    auto issue_tile = [&](int buf, int k0) {
        unsigned add = (unsigned)buf * (unsigned)BUF_BYTES;
        int kb = k0 * 2;
#pragma unroll
        for (int it = 0; it < 2; ++it)
            async_b128(ldsA0[it] + add, voffA[it] + kb, A);
#pragma unroll
        for (int it = 0; it < 2; ++it)
            async_b128(ldsB0[it] + add, voffB[it] + kb, Bt);
    };

    v8f acc[2][4];
#pragma unroll
    for (int i = 0; i < 2; ++i)
#pragma unroll
        for (int j = 0; j < 4; ++j) acc[i][j] = v8f_zero();

    issue_tile(0, 0);

    for (int k0 = 0; k0 < Kp; k0 += GEMM_BK) {
        const int cur = (k0 >> 5) & 1;
        wait_async0();        // this wave's async writes have landed in LDS
        __syncthreads();      // everyone's landed; prev reads of 'cur^1' done
        int kn = k0 + GEMM_BK;
        if (kn < Kp) issue_tile(cur ^ 1, kn);

        // fragments per ISA layout (16-bit A 16x32 / B 32x16, wave32)
        v16bf afrag[2], bfrag[4];
#pragma unroll
        for (int mi = 0; mi < 2; ++mi) {
            int row = wm * 32 + mi * 16 + l15;
            int ko  = half * 8;
            v8bf lo = *(v8bf*)&As[cur][row][ko];
            v8bf hi = *(v8bf*)&As[cur][row][16 + ko];
#pragma unroll
            for (int e = 0; e < 8; ++e) {
                afrag[mi][e]     = lo[e];
                afrag[mi][8 + e] = hi[e];
            }
        }
#pragma unroll
        for (int ni = 0; ni < 4; ++ni) {
            int row = wn * 64 + ni * 16 + l15;
            int ko  = half * 8;
            v8bf lo = *(v8bf*)&Bs[cur][row][ko];
            v8bf hi = *(v8bf*)&Bs[cur][row][16 + ko];
#pragma unroll
            for (int e = 0; e < 8; ++e) {
                bfrag[ni][e]     = lo[e];
                bfrag[ni][8 + e] = hi[e];
            }
        }

#pragma unroll
        for (int mi = 0; mi < 2; ++mi)
#pragma unroll
            for (int ni = 0; ni < 4; ++ni)
                acc[mi][ni] = __builtin_amdgcn_wmma_f32_16x16x32_bf16(
                    false, afrag[mi], false, bfrag[ni],
                    (short)0, acc[mi][ni], false, false);
    }

    // store: C/D layout -> element v of acc is (m = v + 8*half, n = l15)
#pragma unroll
    for (int mi = 0; mi < 2; ++mi) {
#pragma unroll
        for (int ni = 0; ni < 4; ++ni) {
            int colBase = bn0 + wn * 64 + ni * 16 + l15;
#pragma unroll
            for (int v = 0; v < 8; ++v) {
                int row = bm0 + wm * 32 + mi * 16 + half * 8 + v;
                if (row < M) {
                    float val;
                    if (colBase < Nvalid) {
                        val = acc[mi][ni][v];
                        if (bias) val += bias[colBase];
                    } else {
                        val = 0.0f;  // keep pad columns zeroed
                    }
                    C[(size_t)row * ldc + colBase] = val;
                }
            }
        }
    }
}

// ---------------------------------------------------------------------------
// GAT edge kernels (E edges + N self-loops)
// ---------------------------------------------------------------------------
DEV_INLINE void edge_sd(const long long* __restrict__ ei, int e, int E, int& s, int& d) {
    if (e < E) { s = (int)ei[e]; d = (int)ei[(size_t)E + e]; }
    else       { s = d = e - E; }
}

__global__ void k_edge_logits(const long long* __restrict__ ei,
                              const float* __restrict__ als, const float* __restrict__ ald,
                              float* __restrict__ ebuf, unsigned* __restrict__ segmaxk,
                              int E, int Nn) {
    int e = blockIdx.x * blockDim.x + threadIdx.x;
    if (e >= E + Nn) return;
    int s, d;
    edge_sd(ei, e, E, s, d);
    float t = als[s] + ald[d];
    t = (t > 0.0f) ? t : 0.2f * t;  // LeakyReLU(0.2)
    ebuf[e] = t;
    atomicMax(&segmaxk[d], fkey(t));
}

__global__ void k_edge_exp(const long long* __restrict__ ei,
                           float* __restrict__ ebuf, const unsigned* __restrict__ segmaxk,
                           float* __restrict__ segsum, int E, int Nn) {
    int e = blockIdx.x * blockDim.x + threadIdx.x;
    if (e >= E + Nn) return;
    int s, d;
    edge_sd(ei, e, E, s, d);
    float m  = unfkey(segmaxk[d]);
    float ee = __expf(ebuf[e] - m);
    ebuf[e]  = ee;
    atomicAdd(&segsum[d], ee);
}

// one block per edge; threads stride feature dim
__global__ void k_edge_scatter(const long long* __restrict__ ei,
                               const float* __restrict__ ebuf, const float* __restrict__ segsum,
                               const float* __restrict__ xs, float* __restrict__ out,
                               int E, int Nn, int F) {
    int e = blockIdx.x;
    int s, d;
    edge_sd(ei, e, E, s, d);
    float alpha = ebuf[e] / (segsum[d] + 1e-16f);
    const float* srow = xs + (size_t)s * F;
    float* drow = out + (size_t)d * F;
    for (int f = threadIdx.x; f < F; f += blockDim.x)
        atomicAdd(&drow[f], alpha * srow[f]);
}

__global__ void k_bias_relu_cvt(const float* __restrict__ h, const float* __restrict__ b,
                                __bf16* __restrict__ o, size_t tot, int F) {
    size_t i = (size_t)blockIdx.x * blockDim.x + threadIdx.x;
    size_t st = (size_t)gridDim.x * blockDim.x;
    for (; i < tot; i += st) {
        float v = h[i] + b[(int)(i % F)];
        o[i] = (__bf16)(v > 0.0f ? v : 0.0f);
    }
}

// ---------------------------------------------------------------------------
// BatchNorm (training-forward) kernels
// ---------------------------------------------------------------------------
__global__ void k_colstats(const float* __restrict__ X, float* __restrict__ mean,
                           float* __restrict__ rstd, int M, int Nvalid, int ldc) {
    int c = blockIdx.x * blockDim.x + threadIdx.x;
    if (c >= Nvalid) return;
    float s = 0.0f, s2 = 0.0f;
    for (int r = 0; r < M; ++r) {
        float v = X[(size_t)r * ldc + c];
        s  += v;
        s2 += v * v;
    }
    float mu  = s / (float)M;
    float var = s2 / (float)M - mu * mu;
    mean[c] = mu;
    rstd[c] = rsqrtf(var + 1e-5f);
}

__global__ void k_bn_relu_cvt(const float* __restrict__ X,
                              const float* __restrict__ mean, const float* __restrict__ rstd,
                              const float* __restrict__ g, const float* __restrict__ b,
                              __bf16* __restrict__ o, int M, int Nvalid, int ldc, int ldo) {
    size_t tot = (size_t)M * ldo;
    size_t i = (size_t)blockIdx.x * blockDim.x + threadIdx.x;
    size_t st = (size_t)gridDim.x * blockDim.x;
    for (; i < tot; i += st) {
        int r = (int)(i / ldo);
        int c = (int)(i % ldo);
        float v = 0.0f;
        if (c < Nvalid) {
            v = g[c] * (X[(size_t)r * ldc + c] - mean[c]) * rstd[c] + b[c];
            v = v > 0.0f ? v : 0.0f;
        }
        o[i] = (__bf16)v;
    }
}

// ---------------------------------------------------------------------------
// fc4 (256 -> 10) + log_softmax, one thread per node
// ---------------------------------------------------------------------------
__global__ void k_fc4_logsoftmax(const __bf16* __restrict__ X, const float* __restrict__ W,
                                 const float* __restrict__ b, float* __restrict__ out, int Nn) {
    int n = blockIdx.x * blockDim.x + threadIdx.x;
    if (n >= Nn) return;
    float acc[10];
#pragma unroll
    for (int j = 0; j < 10; ++j) acc[j] = b[j];
    const __bf16* row = X + (size_t)n * 256;
    for (int k = 0; k < 256; ++k) {
        float x = (float)row[k];
#pragma unroll
        for (int j = 0; j < 10; ++j) acc[j] += x * W[k * 10 + j];
    }
    float m = acc[0];
#pragma unroll
    for (int j = 1; j < 10; ++j) m = fmaxf(m, acc[j]);
    float s = 0.0f;
#pragma unroll
    for (int j = 0; j < 10; ++j) s += __expf(acc[j] - m);
    float ls = __logf(s);
#pragma unroll
    for (int j = 0; j < 10; ++j) out[(size_t)n * 10 + j] = acc[j] - m - ls;
}

// ---------------------------------------------------------------------------
// Host launch
// ---------------------------------------------------------------------------
static inline int gs_blocks(size_t n, int cap = 8192) {
    size_t b = (n + 255) / 256;
    return (int)(b < (size_t)cap ? b : (size_t)cap);
}

extern "C" void kernel_launch(void* const* d_in, const int* in_sizes, int n_in,
                              void* d_out, int out_size, void* d_ws, size_t ws_size,
                              hipStream_t stream) {
    (void)n_in; (void)out_size; (void)ws_size;

    const int D = 256, H1 = 256, H2 = 1024, F1 = 4092, F1p = 4096;
    const int Nn = in_sizes[0] / D;
    const int E  = in_sizes[1] / 2;
    const int Etot = E + Nn;

    const float*     x      = (const float*)d_in[0];
    const long long* ei     = (const long long*)d_in[1];
    const float*     gc1_Ws = (const float*)d_in[2];
    const float*     gc1_Wd = (const float*)d_in[3];
    const float*     gc1_as = (const float*)d_in[4];
    const float*     gc1_ad = (const float*)d_in[5];
    const float*     gc1_b  = (const float*)d_in[6];
    const float*     gc2_Ws = (const float*)d_in[7];
    const float*     gc2_Wd = (const float*)d_in[8];
    const float*     gc2_as = (const float*)d_in[9];
    const float*     gc2_ad = (const float*)d_in[10];
    const float*     gc2_b  = (const float*)d_in[11];
    const float*     fc1_W  = (const float*)d_in[12];
    const float*     fc1_b  = (const float*)d_in[13];
    const float*     bn1_g  = (const float*)d_in[14];
    const float*     bn1_b  = (const float*)d_in[15];
    const float*     fc2_W  = (const float*)d_in[16];
    const float*     fc2_b  = (const float*)d_in[17];
    const float*     bn2_g  = (const float*)d_in[18];
    const float*     bn2_b  = (const float*)d_in[19];
    const float*     fc3_W  = (const float*)d_in[20];
    const float*     fc3_b  = (const float*)d_in[21];
    const float*     bn3_g  = (const float*)d_in[22];
    const float*     bn3_b  = (const float*)d_in[23];
    const float*     fc4_W  = (const float*)d_in[24];
    const float*     fc4_b  = (const float*)d_in[25];
    float*           out    = (float*)d_out;

    // ---- workspace bump allocator ----
    char*  ws  = (char*)d_ws;
    size_t off = 0;
    auto alloc = [&](size_t bytes) -> char* {
        char* p = ws + off;
        off += (bytes + 255) & ~(size_t)255;
        return p;
    };

    // persistent bf16 weights (transposed, padded)
    __bf16* ws1t = (__bf16*)alloc((size_t)H1 * D * 2);        // [256][256]
    __bf16* ws2t = (__bf16*)alloc((size_t)H2 * H1 * 2);       // [1024][256]
    __bf16* fc1t = (__bf16*)alloc((size_t)F1p * H2 * 2);      // [4096][1024]
    __bf16* fc2t = (__bf16*)alloc((size_t)H2 * F1p * 2);      // [1024][4096]
    __bf16* fc3t = (__bf16*)alloc((size_t)D * H2 * 2);        // [256][1024]
    float*  wsv1 = (float*)alloc(D * 4);
    float*  wdv1 = (float*)alloc(D * 4);
    float*  wsv2 = (float*)alloc(H1 * 4);
    float*  wdv2 = (float*)alloc(H1 * 4);
    float*    als     = (float*)alloc((size_t)Nn * 4);
    float*    ald     = (float*)alloc((size_t)Nn * 4);
    unsigned* segmaxk = (unsigned*)alloc((size_t)Nn * 4);
    float*    segsum  = (float*)alloc((size_t)Nn * 4);
    float*    ebuf    = (float*)alloc((size_t)Etot * 4);
    float*    mean    = (float*)alloc((size_t)F1p * 4);
    float*    rstd    = (float*)alloc((size_t)F1p * 4);

    // reusable slabs (lifetimes are stream-ordered; see sequence below)
    const size_t szA = (size_t)Nn * F1p * 2;  // max: a1bf  (~164 MB)
    const size_t szB = (size_t)Nn * F1p * 4;  // max: a1    (~328 MB)
    const size_t szC = (size_t)Nn * H2 * 4;   // max: h2/a2 (~82 MB)
    char* slabA = alloc(szA);
    char* slabB = alloc(szB);
    char* slabC = alloc(szC);

    __bf16* xbf  = (__bf16*)slabA;                                  // S0-S1
    float*  xs1  = (float*)(slabA + (size_t)Nn * D * 2 + 256);      // S1-S2
    float*  h1   = (float*)slabC;                                   // S2
    __bf16* h1bf = (__bf16*)slabB;                                  // S2-S3
    float*  xs2  = (float*)slabA;                                   // S3-S4
    float*  h2   = (float*)slabC;                                   // S4
    __bf16* h2bf = (__bf16*)slabA;                                  // S4-S5
    float*  a1   = (float*)slabB;                                   // S5
    __bf16* a1bf = (__bf16*)slabA;                                  // S5-S6
    float*  a2   = (float*)slabC;                                   // S6
    __bf16* a2bf = (__bf16*)slabB;                                  // S6-S7
    float*  a3   = (float*)slabC;                                   // S7
    __bf16* a3bf = (__bf16*)slabA;                                  // S7-S8

    // ---- S0: weight prep ----
    k_wt<<<gs_blocks((size_t)H1 * D), 256, 0, stream>>>(gc1_Ws, ws1t, D, H1, D, H1);
    k_wt<<<gs_blocks((size_t)H2 * H1), 256, 0, stream>>>(gc2_Ws, ws2t, H1, H2, H1, H2);
    k_wt<<<gs_blocks((size_t)F1p * H2), 256, 0, stream>>>(fc1_W, fc1t, H2, F1, H2, F1p);
    k_wt<<<gs_blocks((size_t)H2 * F1p), 256, 0, stream>>>(fc2_W, fc2t, F1, H2, F1p, H2);
    k_wt<<<gs_blocks((size_t)D * H2), 256, 0, stream>>>(fc3_W, fc3t, H2, D, H2, D);
    k_wvec<<<1, 256, 0, stream>>>(gc1_Ws, gc1_as, wsv1, D, H1);
    k_wvec<<<1, 256, 0, stream>>>(gc1_Wd, gc1_ad, wdv1, D, H1);
    k_wvec<<<1, 256, 0, stream>>>(gc2_Ws, gc2_as, wsv2, H1, H2);
    k_wvec<<<1, 256, 0, stream>>>(gc2_Wd, gc2_ad, wdv2, H1, H2);
    k_cvt_bf16<<<gs_blocks((size_t)Nn * D), 256, 0, stream>>>(x, xbf, (size_t)Nn * D);

    const int gridM = (Nn + GEMM_BM - 1) / GEMM_BM;
    const int eb    = (Etot + 255) / 256;

    // ---- S1: GAT layer 1 transform ----
    k_gemm_bf16<<<dim3(H1 / GEMM_BN, gridM), 256, 0, stream>>>(
        xbf, ws1t, xs1, nullptr, Nn, H1, D, H1);
    k_rowdot2<<<(Nn + 255) / 256, 256, 0, stream>>>(xbf, wsv1, wdv1, als, ald, Nn, D);

    // ---- S2: GAT layer 1 attention + aggregate ----
    k_fill_u32<<<gs_blocks(Nn), 256, 0, stream>>>(segmaxk, 0u, (size_t)Nn);
    k_fill_f32<<<gs_blocks(Nn), 256, 0, stream>>>(segsum, 0.0f, (size_t)Nn);
    k_edge_logits<<<eb, 256, 0, stream>>>(ei, als, ald, ebuf, segmaxk, E, Nn);
    k_edge_exp<<<eb, 256, 0, stream>>>(ei, ebuf, segmaxk, segsum, E, Nn);
    k_fill_f32<<<gs_blocks((size_t)Nn * H1), 256, 0, stream>>>(h1, 0.0f, (size_t)Nn * H1);
    k_edge_scatter<<<Etot, 256, 0, stream>>>(ei, ebuf, segsum, xs1, h1, E, Nn, H1);
    k_bias_relu_cvt<<<gs_blocks((size_t)Nn * H1), 256, 0, stream>>>(
        h1, gc1_b, h1bf, (size_t)Nn * H1, H1);

    // ---- S3: GAT layer 2 transform ----
    k_gemm_bf16<<<dim3(H2 / GEMM_BN, gridM), 256, 0, stream>>>(
        h1bf, ws2t, xs2, nullptr, Nn, H2, H1, H2);
    k_rowdot2<<<(Nn + 255) / 256, 256, 0, stream>>>(h1bf, wsv2, wdv2, als, ald, Nn, H1);

    // ---- S4: GAT layer 2 attention + aggregate ----
    k_fill_u32<<<gs_blocks(Nn), 256, 0, stream>>>(segmaxk, 0u, (size_t)Nn);
    k_fill_f32<<<gs_blocks(Nn), 256, 0, stream>>>(segsum, 0.0f, (size_t)Nn);
    k_edge_logits<<<eb, 256, 0, stream>>>(ei, als, ald, ebuf, segmaxk, E, Nn);
    k_edge_exp<<<eb, 256, 0, stream>>>(ei, ebuf, segmaxk, segsum, E, Nn);
    k_fill_f32<<<gs_blocks((size_t)Nn * H2), 256, 0, stream>>>(h2, 0.0f, (size_t)Nn * H2);
    k_edge_scatter<<<Etot, 256, 0, stream>>>(ei, ebuf, segsum, xs2, h2, E, Nn, H2);
    k_bias_relu_cvt<<<gs_blocks((size_t)Nn * H2), 256, 0, stream>>>(
        h2, gc2_b, h2bf, (size_t)Nn * H2, H2);

    // ---- S5: fc1 + BN1 + ReLU ----
    k_gemm_bf16<<<dim3(F1p / GEMM_BN, gridM), 256, 0, stream>>>(
        h2bf, fc1t, a1, fc1_b, Nn, F1, H2, F1p);
    k_colstats<<<(F1 + 255) / 256, 256, 0, stream>>>(a1, mean, rstd, Nn, F1, F1p);
    k_bn_relu_cvt<<<gs_blocks((size_t)Nn * F1p), 256, 0, stream>>>(
        a1, mean, rstd, bn1_g, bn1_b, a1bf, Nn, F1, F1p, F1p);

    // ---- S6: fc2 + BN2 + ReLU ----
    k_gemm_bf16<<<dim3(H2 / GEMM_BN, gridM), 256, 0, stream>>>(
        a1bf, fc2t, a2, fc2_b, Nn, H2, F1p, H2);
    k_colstats<<<(H2 + 255) / 256, 256, 0, stream>>>(a2, mean, rstd, Nn, H2, H2);
    k_bn_relu_cvt<<<gs_blocks((size_t)Nn * H2), 256, 0, stream>>>(
        a2, mean, rstd, bn2_g, bn2_b, a2bf, Nn, H2, H2, H2);

    // ---- S7: fc3 + BN3 + ReLU ----
    k_gemm_bf16<<<dim3(D / GEMM_BN, gridM), 256, 0, stream>>>(
        a2bf, fc3t, a3, fc3_b, Nn, D, H2, D);
    k_colstats<<<(D + 255) / 256, 256, 0, stream>>>(a3, mean, rstd, Nn, D, D);
    k_bn_relu_cvt<<<gs_blocks((size_t)Nn * D), 256, 0, stream>>>(
        a3, mean, rstd, bn3_g, bn3_b, a3bf, Nn, D, D, D);

    // ---- S8: fc4 + log_softmax ----
    k_fc4_logsoftmax<<<(Nn + 255) / 256, 256, 0, stream>>>(a3bf, fc4_W, fc4_b, out, Nn);
}
